// LLMModel_15152644620920
// MI455X (gfx1250) — matrix operand order repaired and checked by reading, not compiled, not run on hardware
//
#include <hip/hip_runtime.h>
#include <hip/hip_bf16.h>
#include <stdint.h>

// Problem constants (match reference)
#define T_TOK 16384      // B*S
#define D_DIM 1024
#define E_NUM 8
#define H_DIM 2752       // 43 * 64
#define HC    64         // H-chunk streamed per iteration
#define MT    32         // token rows per block tile
#define XCH   520        // LDS elements per 512-elem half-row chunk (512 data + 8 pad)

typedef __attribute__((ext_vector_type(16))) __bf16 v16bf;
typedef __attribute__((ext_vector_type(8)))  float  v8f;
typedef __attribute__((ext_vector_type(4)))  unsigned int u32x4;
typedef __attribute__((ext_vector_type(4)))  int  i32x4;
typedef __attribute__((ext_vector_type(8)))  int  i32x8;

#if __has_builtin(__builtin_amdgcn_tensor_load_to_lds) && __has_builtin(__builtin_amdgcn_s_wait_tensorcnt)
#define USE_TDM 1
#else
#define USE_TDM 0
#endif

union Frag {
    uint4 q[2];   // 32 bytes = 16 bf16
    v16bf v;
};

__device__ __forceinline__ unsigned int f2bf(float f) {
    unsigned int u = __float_as_uint(f);
    u += 0x7FFFu + ((u >> 16) & 1u);      // round-to-nearest-even
    return (u >> 16) & 0xFFFFu;
}

// ---------------------------------------------------------------- zero output + counters
__global__ void zero_out(float4* __restrict__ out, int n4, int* __restrict__ cnt) {
    int i = blockIdx.x * blockDim.x + threadIdx.x;
    if (i < n4) out[i] = make_float4(0.f, 0.f, 0.f, 0.f);
    if (blockIdx.x == 0 && threadIdx.x < 16) cnt[threadIdx.x] = 0;
}

// ---------------------------------------------------------------- fp32 -> bf16 (packed)
__global__ void cvt_bf16(const float4* __restrict__ in, uint2* __restrict__ out, long n4) {
    long stride = (long)gridDim.x * blockDim.x;
    for (long i = (long)blockIdx.x * blockDim.x + threadIdx.x; i < n4; i += stride) {
        float4 v = in[i];
        uint2 o;
        o.x = f2bf(v.x) | (f2bf(v.y) << 16);
        o.y = f2bf(v.z) | (f2bf(v.w) << 16);
        out[i] = o;
    }
}

// ---------------------------------------------------------------- router: one wave per token
__global__ __launch_bounds__(256) void router(const float* __restrict__ x,
                                              const float* __restrict__ gw,
                                              int*   __restrict__ cnt,
                                              int*   __restrict__ tlist,
                                              float* __restrict__ wlist) {
    const int lane = threadIdx.x & 31;
    const int wid  = threadIdx.x >> 5;
    const int t    = blockIdx.x * 8 + wid;

    float xr[32];
    const float* xp = x + (size_t)t * D_DIM;
#pragma unroll
    for (int j = 0; j < 32; ++j) xr[j] = xp[lane + 32 * j];

    float logit[E_NUM];
#pragma unroll
    for (int e = 0; e < E_NUM; ++e) {
        const float* gp = gw + e * D_DIM;
        float s = 0.f;
#pragma unroll
        for (int j = 0; j < 32; ++j) s += xr[j] * gp[lane + 32 * j];
#pragma unroll
        for (int off = 16; off > 0; off >>= 1) s += __shfl_xor(s, off, 32);
        logit[e] = s;
    }

    if (lane == 0) {
        float m = logit[0];
#pragma unroll
        for (int e = 1; e < E_NUM; ++e) m = fmaxf(m, logit[e]);
        float p[E_NUM];
#pragma unroll
        for (int e = 0; e < E_NUM; ++e) p[e] = __expf(logit[e] - m);
        int i1 = 0; float p1 = p[0];
#pragma unroll
        for (int e = 1; e < E_NUM; ++e) if (p[e] > p1) { p1 = p[e]; i1 = e; }
        int i2 = (i1 == 0) ? 1 : 0; float p2 = p[i2];
#pragma unroll
        for (int e = 0; e < E_NUM; ++e)
            if (e != i1 && e != ((i1 == 0) ? 1 : 0) && p[e] > p2) { p2 = p[e]; i2 = e; }
        float inv = __builtin_amdgcn_rcpf(p1 + p2 + 1e-20f);
        int pos1 = atomicAdd(&cnt[i1], 1);
        tlist[i1 * T_TOK + pos1] = t;  wlist[i1 * T_TOK + pos1] = p1 * inv;
        int pos2 = atomicAdd(&cnt[i2], 1);
        tlist[i2 * T_TOK + pos2] = t;  wlist[i2 * T_TOK + pos2] = p2 * inv;
    }
}

// ---------------------------------------------------------------- fused SwiGLU FFN, WMMA bf16
// grid = (T/MT, E), block = 256 (8 waves). Streams H in chunks of 64.
// X tile staged by TDM gather (tensor_load_to_lds) when available.
__global__ __launch_bounds__(256) void moe_ffn(
    const unsigned short* __restrict__ xb,   // [T][D] bf16
    const unsigned short* __restrict__ w1b,  // [E][H][D] bf16
    const unsigned short* __restrict__ w2b,  // [E][D][H] bf16
    const unsigned short* __restrict__ w3b,  // [E][H][D] bf16
    const int*   __restrict__ cnt,
    const int*   __restrict__ tlist,
    const float* __restrict__ wlist,
    float* __restrict__ out)                 // [T][D] fp32
{
    const int e    = blockIdx.y;
    const int nTok = cnt[e];
    const int m0   = blockIdx.x * MT;
    if (m0 >= nTok) return;                  // uniform early-out

    // X tile: per row, two 512-elem chunks, each padded by 8 elems (TDM pad layout).
    // element (r, c) lives at Xs[(r*2 + (c>>9))*XCH + (c&511)]
    __shared__ __attribute__((aligned(16))) unsigned short Xs[MT * 2 * XCH];
    __shared__ __attribute__((aligned(16))) unsigned short Hs[MT][HC + 8];
    __shared__ int   tokS[MT];
    __shared__ float wS[MT];

    const int tid  = threadIdx.x;
    const int lane = tid & 31;
    const int wid  = tid >> 5;
    const int lo16 = lane & 15;
    const int hi   = lane >> 4;              // 0/1: which half-wave

    if (tid < MT) {
        int idx = m0 + tid;
        if (idx < nTok) { tokS[tid] = tlist[e * T_TOK + idx]; wS[tid] = wlist[e * T_TOK + idx]; }
        else            { tokS[tid] = 0;                      wS[tid] = 0.f; }
    }
    __syncthreads();

    const int nValid = nTok - m0;

#if USE_TDM
    // ---- Tensor Data Mover: gather up to 2x16 token rows into LDS (16-bit row indices)
    if (wid == 0) {
        const int tok_l = tokS[lane];        // lane i holds token index of row i
        const unsigned ldsBase = (unsigned)(uintptr_t)(&Xs[0]);   // low 32 bits = LDS offset
        const unsigned long long ga = (unsigned long long)(uintptr_t)xb;
        const int v1 = __builtin_amdgcn_readfirstlane(nValid >= 16 ? 16 : nValid);
        const int v2 = __builtin_amdgcn_readfirstlane(nValid >= 32 ? 16 : (nValid > 16 ? nValid - 16 : 0));

        // group1: data_size=2B | pad_enable | pad_interval=7 (256 DW) | pad_amount=3 (4 DW)
        i32x8 g1;
        g1[0] = (1 << 16) | (1 << 20) | (7 << 22) | (3 << 25);
        g1[1] = (int)(D_DIM << 16);          // tensor_dim0[15:0] in [31:16]
        g1[2] = (int)((unsigned)T_TOK << 16);// dim0[31:16]=0 ; tensor_dim1[15:0]=T
        g1[3] = (int)(D_DIM << 16);          // dim1[31:16]=0 ; tile_dim0 = 1024
        g1[4] = v1;                          // tile_dim1 = #valid gather rows
        g1[5] = D_DIM;                       // tensor_dim0_stride low32 = 1024
        g1[6] = 0; g1[7] = 0;

        u32x4 g0;
        g0[0] = 1u | (1u << 31);             // count=1, gather_mode=1, 16-bit indices
        g0[1] = ldsBase;
        g0[2] = (unsigned)(ga & 0xFFFFFFFFu);
        g0[3] = (unsigned)((ga >> 32) & 0x1FFFFFFu) | (2u << 30);   // type=2 ("image")

        const i32x8 g4 = {};                 // VADDR4 slot: unused, zeros

        i32x4 gi2, gi3;
#pragma unroll
        for (int j = 0; j < 4; ++j) {
            gi2[j] = (__builtin_amdgcn_readlane(tok_l, 2 * j)     & 0xFFFF) |
                     (__builtin_amdgcn_readlane(tok_l, 2 * j + 1) << 16);
            gi3[j] = (__builtin_amdgcn_readlane(tok_l, 8 + 2 * j)     & 0xFFFF) |
                     (__builtin_amdgcn_readlane(tok_l, 8 + 2 * j + 1) << 16);
        }
        __builtin_amdgcn_tensor_load_to_lds(g0, g1, gi2, gi3, g4, 0);

        if (v2 > 0) {                        // rows 16..31
            g0[1] = ldsBase + 32u * (XCH * 2u);   // 32 chunks * 1040B
            g1[4] = v2;
#pragma unroll
            for (int j = 0; j < 4; ++j) {
                gi2[j] = (__builtin_amdgcn_readlane(tok_l, 16 + 2 * j)     & 0xFFFF) |
                         (__builtin_amdgcn_readlane(tok_l, 16 + 2 * j + 1) << 16);
                gi3[j] = (__builtin_amdgcn_readlane(tok_l, 24 + 2 * j)     & 0xFFFF) |
                         (__builtin_amdgcn_readlane(tok_l, 24 + 2 * j + 1) << 16);
            }
            __builtin_amdgcn_tensor_load_to_lds(g0, g1, gi2, gi3, g4, 0);
        }
        __builtin_amdgcn_s_wait_tensorcnt(0);
    }
    __syncthreads();
#else
    // ---- fallback: cooperative gather into the same chunked-padded layout
    for (int i = tid; i < MT * 2 * 64; i += 256) {       // 64 uint4 per 512-elem chunk
        int ch = i >> 6, q = i & 63;
        int r = ch >> 1, half = ch & 1;
        const uint4* src = (const uint4*)(xb + (size_t)tokS[r] * D_DIM + half * 512);
        *(uint4*)&Xs[ch * XCH + q * 8] = src[q];
    }
    __syncthreads();
#endif

    // phase-A subtile ownership: 2 (m) x 4 (n) tiles cover 32x64
    const int mbA = wid >> 2;                // 0..1
    const int nbA = wid & 3;                 // 0..3
    const int xr  = mbA * 16 + lo16;         // A-frag token row

    // hoisted per-half LDS base pointers: all A-frag ds_loads become base+imm
    const unsigned short* xh0 = &Xs[(xr * 2 + 0) * XCH + hi * 16];
    const unsigned short* xh1 = &Xs[(xr * 2 + 1) * XCH + hi * 16];

    // phase-B accumulators: this wave owns D columns [wid*128, wid*128+128)
    v8f acc[2][8];
#pragma unroll
    for (int a = 0; a < 2; ++a)
#pragma unroll
        for (int b = 0; b < 8; ++b) acc[a][b] = (v8f){};

    const unsigned short* w1e = w1b + (size_t)e * H_DIM * D_DIM;
    const unsigned short* w3e = w3b + (size_t)e * H_DIM * D_DIM;
    const unsigned short* w2e = w2b + (size_t)e * D_DIM * H_DIM;

    // per-lane weight base pointers (byte addressed), advanced by constants per chunk
    const char* pw1  = (const char*)(w1e + (size_t)(nbA * 16 + lo16) * D_DIM + hi * 16);
    const char* pw3  = (const char*)(w3e + (size_t)(nbA * 16 + lo16) * D_DIM + hi * 16);
    const char* pw2c = (const char*)(w2e + (size_t)(wid * 128 + lo16) * H_DIM + hi * 16);

    for (int hc = 0; hc < H_DIM; hc += HC) {
        // ---------- phase A: g1 = x@w1c^T, g3 = x@w3c^T (16x16 tile each), K = D
        v8f g1 = (v8f){}, g3 = (v8f){};
#pragma unroll
        for (int half = 0; half < 2; ++half) {
            const unsigned short* xh = half ? xh1 : xh0;
            const char* ph1 = pw1 + half * 1024;   // 512 elems * 2B
            const char* ph3 = pw3 + half * 1024;
#pragma unroll 4
            for (int kc = 0; kc < 512; kc += 32) {
                Frag a, b1, b3;
                a.q[0] = *(const uint4*)&xh[kc];        // k {0..7 | 16..23} by hi
                a.q[1] = *(const uint4*)&xh[kc + 8];    // k {8..15 | 24..31}
                const uint4* p1 = (const uint4*)(ph1 + kc * 2);
                const uint4* p3 = (const uint4*)(ph3 + kc * 2);
                b1.q[0] = p1[0]; b1.q[1] = p1[1];       // B layout: k 0..15 / 16..31 contiguous
                b3.q[0] = p3[0]; b3.q[1] = p3[1];
                g1 = __builtin_amdgcn_wmma_f32_16x16x32_bf16(false, a.v, false, b1.v, (short)0, g1, false, false);
                g3 = __builtin_amdgcn_wmma_f32_16x16x32_bf16(false, a.v, false, b3.v, (short)0, g3, false, false);
            }
        }
        // h = silu(g1) * g3 -> LDS (bf16); v_rcp instead of IEEE divide
#pragma unroll
        for (int j = 0; j < 8; ++j) {
            float g = g1[j];
            float h = g * __builtin_amdgcn_rcpf(1.f + __expf(-g)) * g3[j];
            Hs[mbA * 16 + j + hi * 8][nbA * 16 + lo16] = (unsigned short)f2bf(h);
        }
        __syncthreads();

        // ---------- phase B: y += h_chunk @ w2c^T for this wave's 128 D-columns
#pragma unroll
        for (int kc2 = 0; kc2 < HC; kc2 += 32) {
            Frag ha[2];
#pragma unroll
            for (int mb = 0; mb < 2; ++mb) {
                int hr = mb * 16 + lo16;
                int ko = kc2 + hi * 16;
                ha[mb].q[0] = *(const uint4*)&Hs[hr][ko];
                ha[mb].q[1] = *(const uint4*)&Hs[hr][ko + 8];
            }
#pragma unroll
            for (int nb = 0; nb < 8; ++nb) {
                Frag b2;
                const uint4* p2 = (const uint4*)(pw2c + (nb * 16 * H_DIM + kc2) * 2);
                b2.q[0] = p2[0]; b2.q[1] = p2[1];
#pragma unroll
                for (int mb = 0; mb < 2; ++mb)
                    acc[mb][nb] = __builtin_amdgcn_wmma_f32_16x16x32_bf16(
                        false, ha[mb].v, false, b2.v, (short)0, acc[mb][nb], false, false);
            }
        }
        __syncthreads();   // before Hs is overwritten next chunk

        pw1  += HC * D_DIM * 2;   // next 64 h-rows of w1/w3
        pw3  += HC * D_DIM * 2;
        pw2c += HC * 2;           // next 64 h-columns of w2
    }

    // ---------- epilogue: scale by routing weight, scatter-add (2 adds/elem -> order-independent)
#pragma unroll
    for (int mb = 0; mb < 2; ++mb)
#pragma unroll
        for (int nb = 0; nb < 8; ++nb) {
            int dcol = wid * 128 + nb * 16 + lo16;
#pragma unroll
            for (int j = 0; j < 8; ++j) {
                int m = mb * 16 + j + hi * 8;
                if (m < nValid) {
                    float v = acc[mb][nb][j] * wS[m];
                    atomicAdd(&out[(size_t)tokS[m] * D_DIM + dcol], v);
                }
            }
        }
}

// ---------------------------------------------------------------- host launcher
extern "C" void kernel_launch(void* const* d_in, const int* in_sizes, int n_in,
                              void* d_out, int out_size, void* d_ws, size_t ws_size,
                              hipStream_t stream) {
    const float* x  = (const float*)d_in[0];   // [B,S,D]
    const float* gw = (const float*)d_in[1];   // [E,D]
    const float* w1 = (const float*)d_in[2];   // [E,H,D]
    const float* w2 = (const float*)d_in[3];   // [E,D,H]
    const float* w3 = (const float*)d_in[4];   // [E,H,D]
    float* out = (float*)d_out;                // [B,S,D] fp32

    // workspace layout (~162 MB): bf16 copies of x/w1/w2/w3 + routing lists
    char* ws = (char*)d_ws;
    size_t off = 0;
    unsigned short* xb  = (unsigned short*)(ws + off); off += (size_t)T_TOK * D_DIM * 2;
    unsigned short* w1b = (unsigned short*)(ws + off); off += (size_t)E_NUM * H_DIM * D_DIM * 2;
    unsigned short* w2b = (unsigned short*)(ws + off); off += (size_t)E_NUM * H_DIM * D_DIM * 2;
    unsigned short* w3b = (unsigned short*)(ws + off); off += (size_t)E_NUM * H_DIM * D_DIM * 2;
    int*   cnt   = (int*)(ws + off);   off += 64;
    int*   tlist = (int*)(ws + off);   off += (size_t)E_NUM * T_TOK * 4;
    float* wlist = (float*)(ws + off); off += (size_t)E_NUM * T_TOK * 4;

    const long nX4 = (long)T_TOK * D_DIM / 4;
    const long nW4 = (long)E_NUM * H_DIM * D_DIM / 4;

    zero_out<<<(int)((nX4 + 255) / 256), 256, 0, stream>>>((float4*)out, (int)nX4, cnt);

    cvt_bf16<<<4096, 256, 0, stream>>>((const float4*)x,  (uint2*)xb,  nX4);
    cvt_bf16<<<8192, 256, 0, stream>>>((const float4*)w1, (uint2*)w1b, nW4);
    cvt_bf16<<<8192, 256, 0, stream>>>((const float4*)w2, (uint2*)w2b, nW4);
    cvt_bf16<<<8192, 256, 0, stream>>>((const float4*)w3, (uint2*)w3b, nW4);

    router<<<T_TOK / 8, 256, 0, stream>>>(x, gw, cnt, tlist, wlist);

    dim3 grid(T_TOK / MT, E_NUM);
    moe_ffn<<<grid, 256, 0, stream>>>(xb, w1b, w2b, w3b, cnt, tlist, wlist, out);
}
